// CrossAttention_48816598286941
// MI455X (gfx1250) — compile-verified, matrix operand order checked by reading, not run on hardware
//
#include <hip/hip_runtime.h>
#include <hip/hip_bf16.h>

// ---------------- constants (match reference) ----------------
#define DIMX 512
#define HEADS 8
#define DH 64
#define BATCH 8
#define LCTX 77
#define CTXD 768
#define LP 80    // L padded to 5*16 for score N-tiles
#define LP2 96   // L padded to 3*32 for PV K-steps
#define NPER 16384
#define NTOT (BATCH * NPER)
#define TM 64        // query rows per workgroup
#define THREADS 256  // 8 wave32
#define SCALE 0.125f // 1/sqrt(64)
#define EPS 1e-5f

typedef __bf16 bf16_t;
typedef __attribute__((ext_vector_type(16))) __bf16 v16bf;
typedef __attribute__((ext_vector_type(8)))  __bf16 v8bf;
typedef __attribute__((ext_vector_type(8)))  float  v8f;

union FragU { v16bf v; v8bf h[2]; };

// 16-bit WMMA fragment from a row-major tile (A: row = M; B-transposed: row = N).
// ISA 7.12.2: lane(l) needs K = half*8 + [0..8) and 16 + half*8 + [0..8),
// i.e. two contiguous 16-byte runs -> 2x b128 loads.
__device__ __forceinline__ v16bf load_frag(const bf16_t* p, int ld, int lane) {
  int row = lane & 15, half = lane >> 4;
  const bf16_t* r = p + row * ld + half * 8;
  FragU u;
  u.h[0] = *(const v8bf*)(r);
  u.h[1] = *(const v8bf*)(r + 16);
  return u.v;
}

__device__ __forceinline__ v8f wmma_bf16(v16bf a, v16bf b, v8f c) {
  return __builtin_amdgcn_wmma_f32_16x16x32_bf16(false, a, false, b, (short)0, c,
                                                 false, false);
}

// ---------------- prep: Wq/Wo -> bf16 (keep original [e][d] = B^T layout) ----------------
__global__ void prep_w_kernel(const float* __restrict__ Wq,
                              const float* __restrict__ Wo,
                              bf16_t* __restrict__ Wqb, bf16_t* __restrict__ Wob) {
  int idx = blockIdx.x * blockDim.x + threadIdx.x;
  if (idx >= 2 * DIMX * DIMX) return;
  int which = idx / (DIMX * DIMX);
  int r = idx - which * (DIMX * DIMX);
  if (which == 0) Wqb[r] = (bf16_t)Wq[r];
  else            Wob[r] = (bf16_t)Wo[r];
}

// ---------------- prep: K/V projection (tiny) ----------------
// Kh[b][h][LP][DH]   (natural layout = B^T for score GEMM, zero-padded l>=77)
// Vt[b][h][DH][LP2]  (transposed    = B^T for PV GEMM,    zero-padded l>=77)
__global__ void prep_kv_kernel(const float* __restrict__ ctx,
                               const float* __restrict__ Wk,
                               const float* __restrict__ Wv,
                               const float* __restrict__ bk,
                               const float* __restrict__ bv,
                               bf16_t* __restrict__ Kh, bf16_t* __restrict__ Vt) {
  int idx = blockIdx.x * blockDim.x + threadIdx.x;
  if (idx >= BATCH * LP2 * DIMX) return;
  int e = idx % DIMX;
  int l = (idx / DIMX) % LP2;
  int b = idx / (DIMX * LP2);
  int h = e / DH, dh = e % DH;
  float kv = 0.f, vv = 0.f;
  if (l < LCTX) {
    const float* c  = ctx + ((size_t)b * LCTX + l) * CTXD;
    const float* wk = Wk + (size_t)e * CTXD;
    const float* wv = Wv + (size_t)e * CTXD;
    for (int i = 0; i < CTXD; ++i) { kv += c[i] * wk[i]; vv += c[i] * wv[i]; }
    kv += bk[e]; vv += bv[e];
  }
  if (l < LP) Kh[(((size_t)b * HEADS + h) * LP + l) * DH + dh] = (bf16_t)kv;
  Vt[(((size_t)b * HEADS + h) * DH + dh) * LP2 + l] = (bf16_t)vv;
}

// ---------------- fused LN + Qproj + attention + Oproj + residual ----------------
// Dynamic LDS (160KB): sQ 64x512 bf16 (64K, x_ln then Q) | sO 64x512 bf16 (64K)
// | 32K scratch: weight panel 512x32 bf16 OR sS 64x80 f32 + sP 64x96 bf16
__global__ __launch_bounds__(THREADS)
void fused_attn_kernel(const float* __restrict__ data,
                       const float* __restrict__ ln_w, const float* __restrict__ ln_b,
                       const float* __restrict__ bq,   const float* __restrict__ bo,
                       const bf16_t* __restrict__ Wqb, const bf16_t* __restrict__ Wob,
                       const bf16_t* __restrict__ Kh,  const bf16_t* __restrict__ Vt,
                       float* __restrict__ out) {
  extern __shared__ char smem[];
  bf16_t* sQ = (bf16_t*)smem;                       // 65536 B
  bf16_t* sO = (bf16_t*)(smem + 65536);             // 65536 B
  char* scratch = smem + 131072;                    // 32768 B
  bf16_t* sPanel = (bf16_t*)scratch;                // [512 rows][32] bf16 (B^T slice)
  float*  sS = (float*)scratch;                     // 64x80 f32 = 20480 B
  bf16_t* sP = (bf16_t*)(scratch + 20480);          // 64x96 bf16 = 12288 B

  const int tid = threadIdx.x;
  const int wave = tid >> 5, lane = tid & 31;
  const int row0 = blockIdx.x * TM;          // contiguous segments: batch = row0/NPER
  const int b = row0 / NPER;
  // 512-wide GEMM mapping: each wave owns a 2x8 block of 16x16 tiles
  const int rowpair = wave & 1;              // rows rowpair*32 .. +32 (2 tiles)
  const int colq = wave >> 1;                // cols colq*128 .. +128 (8 tiles)
  // attention mapping (4x5 and 4x4 tile grids)
  const int rt = wave & 3;
  const int colhalf = wave >> 2;

  // ---- Phase 1: LayerNorm -> bf16 into sQ (lane owns 16 contiguous cols) ----
  for (int r = wave * 8; r < wave * 8 + 8; ++r) {
    const float* xr = data + (size_t)(row0 + r) * DIMX + lane * 16;
    float4 x0 = ((const float4*)xr)[0], x1 = ((const float4*)xr)[1];
    float4 x2 = ((const float4*)xr)[2], x3 = ((const float4*)xr)[3];
    float s  = (x0.x + x0.y + x0.z + x0.w) + (x1.x + x1.y + x1.z + x1.w)
             + (x2.x + x2.y + x2.z + x2.w) + (x3.x + x3.y + x3.z + x3.w);
    float s2 = (x0.x*x0.x + x0.y*x0.y + x0.z*x0.z + x0.w*x0.w)
             + (x1.x*x1.x + x1.y*x1.y + x1.z*x1.z + x1.w*x1.w)
             + (x2.x*x2.x + x2.y*x2.y + x2.z*x2.z + x2.w*x2.w)
             + (x3.x*x3.x + x3.y*x3.y + x3.z*x3.z + x3.w*x3.w);
    for (int off = 16; off >= 1; off >>= 1) {
      s  += __shfl_xor(s,  off, 32);
      s2 += __shfl_xor(s2, off, 32);
    }
    float mu = s * (1.0f / DIMX);
    float rstd = rsqrtf(s2 * (1.0f / DIMX) - mu * mu + EPS);
    const float* wp = ln_w + lane * 16;
    const float* bp = ln_b + lane * 16;
    float xv[16] = {x0.x,x0.y,x0.z,x0.w, x1.x,x1.y,x1.z,x1.w,
                    x2.x,x2.y,x2.z,x2.w, x3.x,x3.y,x3.z,x3.w};
    FragU u;
#pragma unroll
    for (int j = 0; j < 16; ++j)
      u.v[j] = (bf16_t)((xv[j] - mu) * rstd * wp[j] + bp[j]);
    v8bf* dst = (v8bf*)(sQ + r * DIMX + lane * 16);
    dst[0] = u.h[0];
    dst[1] = u.h[1];
  }
  __syncthreads();

  // ---- Phase 2: Q = xln @ Wq^T (64x512 * 512x512; B^T = Wqb row-major) ----
  {
    v8f acc[2][8] = {};
    for (int ks = 0; ks < 16; ++ks) {
      __syncthreads();  // protect panel overwrite
      // stage B^T column slice: panel[n][0..32) = Wqb[n][ks*32 .. +32)
      {
        const bf16_t* src0 = Wqb + (size_t)(tid * 2) * DIMX + ks * 32;
        uint4* d0 = (uint4*)(sPanel + (tid * 2) * 32);
        uint4* d1 = (uint4*)(sPanel + (tid * 2 + 1) * 32);
        const uint4* s0 = (const uint4*)src0;
        const uint4* s1 = (const uint4*)(src0 + DIMX);
        d0[0]=s0[0]; d0[1]=s0[1]; d0[2]=s0[2]; d0[3]=s0[3];
        d1[0]=s1[0]; d1[1]=s1[1]; d1[2]=s1[2]; d1[3]=s1[3];
        if (ks + 1 < 16)
          __builtin_prefetch(Wqb + (size_t)(tid * 2) * DIMX + (ks + 1) * 32, 0, 3);
      }
      __syncthreads();
      v16bf a0 = load_frag(sQ + (rowpair * 32) * DIMX + ks * 32, DIMX, lane);
      v16bf a1 = load_frag(sQ + (rowpair * 32 + 16) * DIMX + ks * 32, DIMX, lane);
#pragma unroll
      for (int ct = 0; ct < 8; ++ct) {
        v16bf bb = load_frag(sPanel + (colq * 128 + ct * 16) * 32, 32, lane);
        acc[0][ct] = wmma_bf16(a0, bb, acc[0][ct]);
        acc[1][ct] = wmma_bf16(a1, bb, acc[1][ct]);
      }
    }
    __syncthreads();  // all waves done reading xln; overwrite sQ with Q
#pragma unroll
    for (int rr = 0; rr < 2; ++rr)
#pragma unroll
      for (int ct = 0; ct < 8; ++ct)
#pragma unroll
        for (int r = 0; r < 8; ++r) {
          int row = rowpair * 32 + rr * 16 + r + 8 * (lane >> 4);
          int col = colq * 128 + ct * 16 + (lane & 15);
          sQ[row * DIMX + col] = (bf16_t)(acc[rr][ct][r] + bq[col]);
        }
    __syncthreads();
  }

  // ---- Phase 3: per-head attention ----
  for (int h = 0; h < HEADS; ++h) {
    // scores S[64][80] = Q_h(64x64) x K_h^T; B^T = Kh[l][dh] natural layout
    const bf16_t* Kth = Kh + ((size_t)b * HEADS + h) * LP * DH;
    const int nct = (wave < 4) ? 3 : 2;
    const int ct0 = (wave < 4) ? 0 : 3;
    v8f sacc[3] = {};
    for (int kk = 0; kk < 2; ++kk) {
      v16bf a = load_frag(sQ + rt * 16 * DIMX + h * DH + kk * 32, DIMX, lane);
      for (int t = 0; t < nct; ++t) {
        v16bf bb = load_frag(Kth + (ct0 + t) * 16 * DH + kk * 32, DH, lane);
        sacc[t] = wmma_bf16(a, bb, sacc[t]);
      }
    }
    for (int t = 0; t < nct; ++t)
#pragma unroll
      for (int r = 0; r < 8; ++r) {
        int row = rt * 16 + r + 8 * (lane >> 4);
        int col = (ct0 + t) * 16 + (lane & 15);
        sS[row * LP + col] = sacc[t][r];
      }
    __syncthreads();

    // softmax over l<77 (one thread per row)
    if (tid < TM) {
      float mx = -1e30f;
      for (int l = 0; l < LCTX; ++l) {
        float v = sS[tid * LP + l] * SCALE;
        mx = v > mx ? v : mx;
      }
      float sum = 0.f;
      for (int l = 0; l < LCTX; ++l) {
        float e = __expf(sS[tid * LP + l] * SCALE - mx);
        sS[tid * LP + l] = e; sum += e;
      }
      float inv = 1.0f / sum;
      for (int l = 0; l < LP2; ++l)
        sP[tid * LP2 + l] = (bf16_t)((l < LCTX) ? sS[tid * LP + l] * inv : 0.0f);
    }
    __syncthreads();

    // O_h(64x64) = P(64x96) x V_h(96x64); B^T = Vt[dh][l]
    const bf16_t* Vth = Vt + ((size_t)b * HEADS + h) * DH * LP2;
    v8f oacc[2] = {};
    for (int kk = 0; kk < 3; ++kk) {
      v16bf a = load_frag(sP + rt * 16 * LP2 + kk * 32, LP2, lane);
#pragma unroll
      for (int t = 0; t < 2; ++t) {
        int ct = colhalf * 2 + t;
        v16bf bb = load_frag(Vth + ct * 16 * LP2 + kk * 32, LP2, lane);
        oacc[t] = wmma_bf16(a, bb, oacc[t]);
      }
    }
#pragma unroll
    for (int t = 0; t < 2; ++t) {
      int ct = colhalf * 2 + t;
#pragma unroll
      for (int r = 0; r < 8; ++r) {
        int row = rt * 16 + r + 8 * (lane >> 4);
        int col = h * DH + ct * 16 + (lane & 15);
        sO[row * DIMX + col] = (bf16_t)oacc[t][r];
      }
    }
    __syncthreads();
  }

  // ---- Phase 4: out = O @ Wo^T + bo + data ----
  {
    v8f acc[2][8] = {};
    for (int ks = 0; ks < 16; ++ks) {
      __syncthreads();
      {
        const bf16_t* src0 = Wob + (size_t)(tid * 2) * DIMX + ks * 32;
        uint4* d0 = (uint4*)(sPanel + (tid * 2) * 32);
        uint4* d1 = (uint4*)(sPanel + (tid * 2 + 1) * 32);
        const uint4* s0 = (const uint4*)src0;
        const uint4* s1 = (const uint4*)(src0 + DIMX);
        d0[0]=s0[0]; d0[1]=s0[1]; d0[2]=s0[2]; d0[3]=s0[3];
        d1[0]=s1[0]; d1[1]=s1[1]; d1[2]=s1[2]; d1[3]=s1[3];
        if (ks + 1 < 16)
          __builtin_prefetch(Wob + (size_t)(tid * 2) * DIMX + (ks + 1) * 32, 0, 3);
      }
      __syncthreads();
      v16bf a0 = load_frag(sO + (rowpair * 32) * DIMX + ks * 32, DIMX, lane);
      v16bf a1 = load_frag(sO + (rowpair * 32 + 16) * DIMX + ks * 32, DIMX, lane);
#pragma unroll
      for (int ct = 0; ct < 8; ++ct) {
        v16bf bb = load_frag(sPanel + (colq * 128 + ct * 16) * 32, 32, lane);
        acc[0][ct] = wmma_bf16(a0, bb, acc[0][ct]);
        acc[1][ct] = wmma_bf16(a1, bb, acc[1][ct]);
      }
    }
#pragma unroll
    for (int rr = 0; rr < 2; ++rr)
#pragma unroll
      for (int ct = 0; ct < 8; ++ct)
#pragma unroll
        for (int r = 0; r < 8; ++r) {
          int row = rowpair * 32 + rr * 16 + r + 8 * (lane >> 4);
          int col = colq * 128 + ct * 16 + (lane & 15);
          size_t gr = (size_t)(row0 + row);
          out[gr * DIMX + col] = acc[rr][ct][r] + bo[col] + data[gr * DIMX + col];
        }
  }
}

// ---------------- launcher ----------------
extern "C" void kernel_launch(void* const* d_in, const int* in_sizes, int n_in,
                              void* d_out, int out_size, void* d_ws, size_t ws_size,
                              hipStream_t stream) {
  const float* data    = (const float*)d_in[0];
  const float* context = (const float*)d_in[1];
  // d_in[2] = batch_id: equal contiguous segments -> implicit (row/NPER)
  const float* ln_w = (const float*)d_in[3];
  const float* ln_b = (const float*)d_in[4];
  const float* Wq   = (const float*)d_in[5];
  const float* Wk   = (const float*)d_in[6];
  const float* Wv   = (const float*)d_in[7];
  const float* bq   = (const float*)d_in[8];
  const float* bk   = (const float*)d_in[9];
  const float* bv   = (const float*)d_in[10];
  const float* Wo   = (const float*)d_in[11];
  const float* bo   = (const float*)d_in[12];
  float* out = (float*)d_out;

  char* ws = (char*)d_ws;
  bf16_t* Wqb = (bf16_t*)(ws);                 // 512KB ([e][d] row-major = B^T)
  bf16_t* Wob = (bf16_t*)(ws + 524288);        // 512KB
  bf16_t* Kh  = (bf16_t*)(ws + 1048576);       // 8*8*80*64*2 = 640KB
  bf16_t* Vt  = (bf16_t*)(ws + 1703936);       // 8*8*64*96*2 = 768KB

  (void)hipFuncSetAttribute((const void*)fused_attn_kernel,
                            hipFuncAttributeMaxDynamicSharedMemorySize, 163840);

  prep_w_kernel<<<(2 * DIMX * DIMX + 255) / 256, 256, 0, stream>>>(Wq, Wo, Wqb, Wob);
  prep_kv_kernel<<<(BATCH * LP2 * DIMX + 255) / 256, 256, 0, stream>>>(
      context, Wk, Wv, bk, bv, Kh, Vt);
  fused_attn_kernel<<<NTOT / TM, THREADS, 163840, stream>>>(
      data, ln_w, ln_b, bq, bo, Wqb, Wob, Kh, Vt, out);
}
